// ModifiedMSA_74680891343410
// MI455X (gfx1250) — matrix operand order repaired and checked by reading, not compiled, not run on hardware
//
#include <hip/hip_runtime.h>
#include <hip/hip_bf16.h>

#define DEVINL __device__ __forceinline__

#define B_ 2
#define H_ 16
#define N_ 4096
#define D_ 64
#define L_ 64
#define E_ 1024
#define SEG_ 64          // N_/L_
#define PART_STRIDE 1056 // 16 max + 16 sum + 16*64 acc (floats)

typedef _Float16 f16_t;
typedef __attribute__((ext_vector_type(16))) _Float16 v16h;
typedef __attribute__((ext_vector_type(8)))  _Float16 v8h;
typedef __attribute__((ext_vector_type(4)))  _Float16 v4h;
typedef __attribute__((ext_vector_type(8)))  float    v8f;
typedef __attribute__((ext_vector_type(4)))  float    v4f;

DEVINL v8f zero8() { v8f z; for (int i = 0; i < 8; ++i) z[i] = 0.0f; return z; }

// D = A*B + C, f16 inputs, f32 accumulate (v_wmma_f32_16x16x32_f16)
DEVINL v8f wmma_f16(v16h a, v16h b, v8f c) {
  return __builtin_amdgcn_wmma_f32_16x16x32_f16(false, a, false, b, (short)0, c,
                                                false, false);
}

// A fragment (16x32 f16, MxK) from row-major MxK tile; ld = row stride in halves.
// Lane l: row M = l&15; halves 0..15 = K (l>=16?16:0)+i (contiguous).
DEVINL v16h load_a_frag(const f16_t* tile, int ld, int k0) {
  int lane = threadIdx.x & 31;
  int row  = lane & 15;
  int kb   = (lane >> 4) << 4;
  const f16_t* r = tile + row * ld + k0 + kb;
  v8h lo = *(const v8h*)(r);
  v8h hi = *(const v8h*)(r + 8);
  v16h a;
  for (int i = 0; i < 8; ++i) { a[i] = lo[i]; a[i + 8] = hi[i]; }
  return a;
}

// B fragment (32x16 f16, KxN) from row-major NxK storage (i.e. B^T rows).
// Lane l: col N = l&15; halves 0..15 = K (l>=16?16:0)+i (contiguous).
DEVINL v16h load_b_frag(const f16_t* tileT, int ld, int k0) {
  int lane = threadIdx.x & 31;
  int row  = lane & 15;
  int kb   = (lane >> 4) << 4;
  return *(const v16h*)(tileT + row * ld + k0 + kb);
}

// reductions across the 16-lane half-groups (masks < 16 stay within a half)
DEVINL float red_max16(float v) {
  for (int m = 1; m < 16; m <<= 1) v = fmaxf(v, __shfl_xor(v, m, 32));
  return v;
}
DEVINL float red_sum16(float v) {
  for (int m = 1; m < 16; m <<= 1) v += __shfl_xor(v, m, 32);
  return v;
}

// ---------------------------------------------------------------- kernel 1
// landmark means: qth = mean(q)/SCALE, kth = mean(k)  (fp16, [BH,L,D])
__global__ void nys_landmarks(const float* __restrict__ Q,
                              const float* __restrict__ K,
                              f16_t* __restrict__ qth, f16_t* __restrict__ kth) {
  int idx = blockIdx.x;               // bh*L + l
  int l   = idx & (L_ - 1);
  int bh  = idx >> 6;
  int b   = bh >> 4, h = bh & 15;
  int d   = threadIdx.x;              // 0..63
  size_t base = ((size_t)b * N_ + (size_t)l * SEG_) * E_ + h * D_ + d;
  float qs = 0.0f, ks = 0.0f;
  for (int s = 0; s < SEG_; ++s) {
    qs += Q[base + (size_t)s * E_];
    ks += K[base + (size_t)s * E_];
  }
  qth[(bh * L_ + l) * D_ + d] = (f16_t)(qs * (1.0f / (SEG_ * 8.0f)));
  kth[(bh * L_ + l) * D_ + d] = (f16_t)(ks * (1.0f / SEG_));
}

// ---------------------------------------------------------------- kernel 2
// kernel2 = softmax(qt @ kt^T) then Gauss-Jordan inverse -> k2inv fp16
__global__ void nys_k2inv(const f16_t* __restrict__ qth,
                          const f16_t* __restrict__ kth,
                          f16_t* __restrict__ k2inv) {
  __shared__ __align__(16) float GJ[L_ * 2 * L_];   // 64 x 128 augmented
  __shared__ float fac[L_];
  __shared__ float pivinv;
  int bh  = blockIdx.x;
  int tid = threadIdx.x;

  if (tid < 32) {                                   // wave 0: WMMA scores
    const f16_t* qt = qth + bh * L_ * D_;
    const f16_t* kt = kth + bh * L_ * D_;
    int grp = tid >> 4, col = tid & 15;
    for (int mt = 0; mt < 4; ++mt)
      for (int nt = 0; nt < 4; ++nt) {
        v8f acc = zero8();
        for (int k0 = 0; k0 < 64; k0 += 32)
          acc = wmma_f16(load_a_frag(qt + mt * 16 * D_, D_, k0),
                         load_b_frag(kt + nt * 16 * D_, D_, k0), acc);
        int n = nt * 16 + col;
        for (int j = 0; j < 8; ++j)
          GJ[(mt * 16 + j + grp * 8) * 128 + n] = acc[j];
      }
  }
  __syncthreads();
  if (tid < 64) {                                   // row softmax + identity
    float* row = GJ + tid * 128;
    float mx = -3.0e38f;
    for (int c = 0; c < 64; ++c) mx = fmaxf(mx, row[c]);
    float s = 0.0f;
    for (int c = 0; c < 64; ++c) { float e = __expf(row[c] - mx); row[c] = e; s += e; }
    float inv = 1.0f / s;
    for (int c = 0; c < 64; ++c) row[c] *= inv;
    for (int c = 0; c < 64; ++c) row[64 + c] = (c == tid) ? 1.0f : 0.0f;
  }
  __syncthreads();
  for (int p = 0; p < 64; ++p) {                    // Gauss-Jordan
    if (tid == 0) pivinv = 1.0f / GJ[p * 128 + p];
    __syncthreads();
    if (tid < 128) GJ[p * 128 + tid] *= pivinv;
    if (tid < 64)  fac[tid] = GJ[tid * 128 + p];
    __syncthreads();
    for (int e = tid; e < 64 * 128; e += 256) {
      int r = e >> 7;
      if (r != p) GJ[e] -= fac[r] * GJ[p * 128 + (e & 127)];
    }
    __syncthreads();
  }
  for (int e = tid; e < 64 * 64; e += 256)
    k2inv[bh * L_ * L_ + e] = (f16_t)GJ[(e >> 6) * 128 + 64 + (e & 63)];
}

// ---------------------------------------------------------------- kernel 3
// partial flash-style softmax(qt @ k^T) @ v over an N-split of 512 keys
__global__ void nys_k3_partial(const float* __restrict__ Kg,
                               const float* __restrict__ Vg,
                               const f16_t* __restrict__ qth,
                               float* __restrict__ partials) {
  __shared__ __align__(32) f16_t Klds[32 * D_];     // keys, row-major [n][d]
  __shared__ __align__(32) f16_t VTlds[D_ * 32];    // V^T, row-major [d][n]
  __shared__ __align__(32) f16_t Plds[16 * 32];     // probs tile [l][n]
  int x  = blockIdx.x;
  int s  = x & 7;
  int rt = (x >> 3) & 3;
  int bh = x >> 5;
  int b = bh >> 4, h = bh & 15;
  int lane = threadIdx.x;
  int grp = lane >> 4, col = lane & 15;

  const f16_t* qt = qth + (bh * L_ + rt * 16) * D_;
  v16h aq0 = load_a_frag(qt, D_, 0);
  v16h aq1 = load_a_frag(qt, D_, 32);

  v8f acc[4];
  for (int dt = 0; dt < 4; ++dt) acc[dt] = zero8();
  float rmax[8], rsum[8];
  for (int j = 0; j < 8; ++j) { rmax[j] = -3.0e38f; rsum[j] = 0.0f; }

  for (int it = 0; it < 16; ++it) {
    int nbase = s * 512 + it * 32;
    size_t gbase = ((size_t)b * N_ + nbase) * E_ + h * D_;
    // stage 32x64 fp32 K,V as fp16 (V transposed) with float4 loads
    for (int q4 = lane; q4 < 32 * 16; q4 += 32) {
      int n = q4 >> 4, d4 = (q4 & 15) << 2;
      size_t g = gbase + (size_t)n * E_ + d4;
      v4f kv = *(const v4f*)(Kg + g);
      v4f vv = *(const v4f*)(Vg + g);
      v4h kh;
      for (int i = 0; i < 4; ++i) kh[i] = (f16_t)kv[i];
      *(v4h*)(Klds + n * D_ + d4) = kh;
      for (int i = 0; i < 4; ++i) VTlds[(d4 + i) * 32 + n] = (f16_t)vv[i];
    }
    v8f s0 = zero8(), s1 = zero8();
    s0 = wmma_f16(aq0, load_b_frag(Klds, D_, 0), s0);
    s0 = wmma_f16(aq1, load_b_frag(Klds, D_, 32), s0);
    s1 = wmma_f16(aq0, load_b_frag(Klds + 16 * D_, D_, 0), s1);
    s1 = wmma_f16(aq1, load_b_frag(Klds + 16 * D_, D_, 32), s1);

    for (int j = 0; j < 8; ++j) {                   // online softmax update
      float tm = red_max16(fmaxf(s0[j], s1[j]));
      float nm = fmaxf(rmax[j], tm);
      float sc = __expf(rmax[j] - nm);
      float p0 = __expf(s0[j] - nm);
      float p1 = __expf(s1[j] - nm);
      float ts = red_sum16(p0 + p1);
      rsum[j] = rsum[j] * sc + ts;
      rmax[j] = nm;
      for (int dt = 0; dt < 4; ++dt) acc[dt][j] *= sc;
      int m = j + (grp << 3);
      Plds[m * 32 + col]      = (f16_t)p0;
      Plds[m * 32 + col + 16] = (f16_t)p1;
    }
    v16h ap = load_a_frag(Plds, 32, 0);             // P [16x32]
    for (int dt = 0; dt < 4; ++dt)
      acc[dt] = wmma_f16(ap, load_b_frag(VTlds + dt * 16 * 32, 32, 0), acc[dt]);
  }

  float* part = partials + (size_t)x * PART_STRIDE;
  if (col == 0)
    for (int j = 0; j < 8; ++j) {
      int m = j + grp * 8;
      part[m] = rmax[j];
      part[16 + m] = rsum[j];
    }
  for (int dt = 0; dt < 4; ++dt)
    for (int j = 0; j < 8; ++j)
      part[32 + (j + grp * 8) * D_ + dt * 16 + col] = acc[dt][j];
}

// log-sum-exp merge of the 8 N-splits -> (K3 @ V)^T as fp16 [BH, D, L]
__global__ void nys_k3_merge(const float* __restrict__ partials,
                             f16_t* __restrict__ k3vT) {
  __shared__ float w[16 * 8];
  __shared__ float inv[16];
  int rt = blockIdx.x & 3;
  int bh = blockIdx.x >> 2;
  int tid = threadIdx.x;                            // 64
  const float* base = partials + (size_t)((bh * 4 + rt) * 8) * PART_STRIDE;
  if (tid < 16) {
    float M = -3.0e38f;
    for (int s = 0; s < 8; ++s) M = fmaxf(M, base[s * PART_STRIDE + tid]);
    float W = 0.0f;
    for (int s = 0; s < 8; ++s) {
      float ws = __expf(base[s * PART_STRIDE + tid] - M);
      w[tid * 8 + s] = ws;
      W += base[s * PART_STRIDE + 16 + tid] * ws;
    }
    inv[tid] = 1.0f / W;
  }
  __syncthreads();
  for (int e4 = tid; e4 < 16 * 16; e4 += 64) {      // float4 over acc rows
    int m = e4 >> 4, d4 = (e4 & 15) << 2;
    v4f v; for (int i = 0; i < 4; ++i) v[i] = 0.0f;
    for (int s = 0; s < 8; ++s) {
      v4f p = *(const v4f*)(base + s * PART_STRIDE + 32 + m * D_ + d4);
      float ws = w[m * 8 + s];
      for (int i = 0; i < 4; ++i) v[i] += p[i] * ws;
    }
    float iv = inv[m];
    for (int i = 0; i < 4; ++i)
      k3vT[bh * L_ * D_ + (d4 + i) * L_ + (rt * 16 + m)] = (f16_t)(v[i] * iv);
  }
}

// ---------------------------------------------------------------- kernel 4
// zv = k2inv @ (K3@V); store transposed fp16 [BH, D, L] as WMMA B-operand
__global__ void nys_zv(const f16_t* __restrict__ k2inv,
                       const f16_t* __restrict__ k3vT,
                       f16_t* __restrict__ zvT) {
  int bh = blockIdx.x;
  int lane = threadIdx.x;
  int grp = lane >> 4, col = lane & 15;
  const f16_t* A  = k2inv + bh * L_ * L_;
  const f16_t* Bt = k3vT + bh * D_ * L_;
  for (int mt = 0; mt < 4; ++mt)
    for (int nt = 0; nt < 4; ++nt) {
      v8f acc = zero8();
      for (int k0 = 0; k0 < 64; k0 += 32)
        acc = wmma_f16(load_a_frag(A + mt * 16 * L_, L_, k0),
                       load_b_frag(Bt + nt * 16 * L_, L_, k0), acc);
      int d = nt * 16 + col;
      for (int j = 0; j < 8; ++j)
        zvT[bh * D_ * L_ + d * L_ + (mt * 16 + j + grp * 8)] = (f16_t)acc[j];
    }
}

// ---------------------------------------------------------------- kernel 5
// out = softmax(q @ kt^T) @ zv, one 16-query tile per wave
__global__ void nys_out(const float* __restrict__ Qg,
                        const f16_t* __restrict__ kth,
                        const f16_t* __restrict__ zvT,
                        float* __restrict__ Og) {
  __shared__ __align__(32) f16_t Qlds[16 * D_];
  __shared__ __align__(32) f16_t Plds[16 * L_];
  int nt = blockIdx.x & 255;
  int bh = blockIdx.x >> 8;
  int b = bh >> 4, h = bh & 15;
  int lane = threadIdx.x;
  int grp = lane >> 4, col = lane & 15;
  int nbase = nt * 16;

  size_t gq = ((size_t)b * N_ + nbase) * E_ + h * D_;
  for (int q4 = lane; q4 < 16 * 16; q4 += 32) {     // float4 Q staging
    int n = q4 >> 4, d4 = (q4 & 15) << 2;
    v4f qv = *(const v4f*)(Qg + gq + (size_t)n * E_ + d4);
    v4h qh;
    for (int i = 0; i < 4; ++i) qh[i] = (f16_t)(qv[i] * 0.125f);
    *(v4h*)(Qlds + n * D_ + d4) = qh;
  }
  const f16_t* kt = kth + bh * L_ * D_;
  v16h aq0 = load_a_frag(Qlds, D_, 0);
  v16h aq1 = load_a_frag(Qlds, D_, 32);
  v8f sacc[4];
  for (int lt = 0; lt < 4; ++lt) {
    sacc[lt] = zero8();
    sacc[lt] = wmma_f16(aq0, load_b_frag(kt + lt * 16 * D_, D_, 0), sacc[lt]);
    sacc[lt] = wmma_f16(aq1, load_b_frag(kt + lt * 16 * D_, D_, 32), sacc[lt]);
  }
  for (int j = 0; j < 8; ++j) {                     // full-row softmax (L=64)
    float mx = fmaxf(fmaxf(sacc[0][j], sacc[1][j]), fmaxf(sacc[2][j], sacc[3][j]));
    mx = red_max16(mx);
    float p[4], sum = 0.0f;
    for (int lt = 0; lt < 4; ++lt) { p[lt] = __expf(sacc[lt][j] - mx); sum += p[lt]; }
    sum = red_sum16(sum);
    float inv = 1.0f / sum;
    int m = j + grp * 8;
    for (int lt = 0; lt < 4; ++lt)
      Plds[m * L_ + lt * 16 + col] = (f16_t)(p[lt] * inv);
  }
  v16h ap0 = load_a_frag(Plds, L_, 0);
  v16h ap1 = load_a_frag(Plds, L_, 32);
  const f16_t* zt = zvT + bh * D_ * L_;
  float* og = Og + ((size_t)b * N_ + nbase) * E_ + h * D_;
  for (int dt = 0; dt < 4; ++dt) {
    v8f o = zero8();
    o = wmma_f16(ap0, load_b_frag(zt + dt * 16 * L_, L_, 0), o);
    o = wmma_f16(ap1, load_b_frag(zt + dt * 16 * L_, L_, 32), o);
    for (int j = 0; j < 8; ++j)
      og[(size_t)(j + grp * 8) * E_ + dt * 16 + col] = o[j];
  }
}

extern "C" void kernel_launch(void* const* d_in, const int* in_sizes, int n_in,
                              void* d_out, int out_size, void* d_ws, size_t ws_size,
                              hipStream_t stream) {
  (void)in_sizes; (void)n_in; (void)out_size; (void)ws_size;
  const float* Q = (const float*)d_in[0];
  const float* K = (const float*)d_in[1];
  const float* V = (const float*)d_in[2];
  float* out = (float*)d_out;

  char* ws = (char*)d_ws;
  size_t o = 0;
  const size_t LD = (size_t)B_ * H_ * L_ * D_;      // 131072 elements
  f16_t* qth   = (f16_t*)(ws + o); o += LD * sizeof(f16_t);
  f16_t* kth   = (f16_t*)(ws + o); o += LD * sizeof(f16_t);
  f16_t* k2inv = (f16_t*)(ws + o); o += LD * sizeof(f16_t);
  f16_t* k3vT  = (f16_t*)(ws + o); o += LD * sizeof(f16_t);
  f16_t* zvT   = (f16_t*)(ws + o); o += LD * sizeof(f16_t);
  float* partials = (float*)(ws + o); // B*H*4*8*PART_STRIDE floats (~4.3 MB)

  nys_landmarks<<<B_ * H_ * L_, 64, 0, stream>>>(Q, K, qth, kth);
  nys_k2inv<<<B_ * H_, 256, 0, stream>>>(qth, kth, k2inv);
  nys_k3_partial<<<B_ * H_ * 32, 32, 0, stream>>>(K, V, qth, partials);
  nys_k3_merge<<<B_ * H_ * 4, 64, 0, stream>>>(partials, k3vT);
  nys_zv<<<B_ * H_, 32, 0, stream>>>(k2inv, k3vT, zvT);
  nys_out<<<B_ * H_ * 256, 32, 0, stream>>>(Q, kth, zvT, out);
}
